// DecoderWithAttention_11424613007890
// MI455X (gfx1250) — compile-verified, matrix operand order checked by reading, not run on hardware
//
#include <hip/hip_runtime.h>

#define B_      64
#define P_      196
#define ENC_    512
#define DEC_    512
#define EMB_    256
#define ATT_    128
#define FFN_    1024
#define V_      2000
#define MAXLEN_ 160
#define T_      159
#define EPS_    1e-5f

typedef __attribute__((ext_vector_type(16))) __bf16 v16bf;
typedef __attribute__((ext_vector_type(8)))  __bf16 v8bf;
typedef __attribute__((ext_vector_type(8)))  float  v8f;
typedef __attribute__((ext_vector_type(4)))  float  v4f;

// ---------------------------------------------------------------------------
// Skinny GEMM: C[M x N] = A[M x K](f32) * W[N x K](bf16)^T + bias
// Optional relu and per-row time mask. One wave per 16x16 C tile, 4 waves
// (4 N-tiles) per 128-thread block. v_wmma_f32_16x16x32_bf16, f32 accumulate.
// Branch-free inner loop: OOB rows are CLAMPED (their garbage only reaches
// C rows/cols that are never stored), loads are 16-byte vectorized.
// Requires K % 64 == 0 (true for all call sites: 512, 768, 1024).
// ---------------------------------------------------------------------------
__device__ __forceinline__ v16bf load_a16(const float* __restrict__ p, int ka, int kb)
{
    const v4f a0 = *(const v4f*)(p + ka);
    const v4f a1 = *(const v4f*)(p + ka + 4);
    const v4f a2 = *(const v4f*)(p + kb);
    const v4f a3 = *(const v4f*)(p + kb + 4);
    const v8f lo = __builtin_shufflevector(a0, a1, 0, 1, 2, 3, 4, 5, 6, 7);
    const v8f hi = __builtin_shufflevector(a2, a3, 0, 1, 2, 3, 4, 5, 6, 7);
    const v8bf blo = __builtin_convertvector(lo, v8bf);
    const v8bf bhi = __builtin_convertvector(hi, v8bf);
    return __builtin_shufflevector(blo, bhi, 0, 1, 2, 3, 4, 5, 6, 7,
                                   8, 9, 10, 11, 12, 13, 14, 15);
}

__device__ __forceinline__ v16bf load_b16(const __bf16* __restrict__ p, int ka, int kb)
{
    const v8bf b0 = *(const v8bf*)(p + ka);
    const v8bf b1 = *(const v8bf*)(p + kb);
    return __builtin_shufflevector(b0, b1, 0, 1, 2, 3, 4, 5, 6, 7,
                                   8, 9, 10, 11, 12, 13, 14, 15);
}

__global__ void gemm_bf16_wmma(const float* __restrict__ A, int M, int lda,
                               const __bf16* __restrict__ W, int N, int K,
                               float* __restrict__ C, long long ldc,
                               const float* __restrict__ bias, int relu,
                               const int* __restrict__ dec_lens, int t)
{
    const int wave  = threadIdx.x >> 5;          // 0..3
    const int lane  = threadIdx.x & 31;
    const int ntile = blockIdx.x * 4 + wave;
    const int mtile = blockIdx.y;
    const int n0 = ntile * 16, m0 = mtile * 16;
    if (n0 >= N) return;                         // uniform per wave

    const int half = lane >> 4;                  // 0: K {0..7,16..23}; 1: K {8..15,24..31}
    const int l    = lane & 15;
    // Clamp OOB rows in-bounds: their products land only in C rows/cols >= M/N,
    // which are never stored below, so correctness is unaffected.
    const int arow = min(m0 + l, M - 1);
    const int wrow = min(n0 + l, N - 1);
    const float*  Ap = A + (long long)arow * lda;
    const __bf16* Wp = W + (long long)wrow * K;
    const int koff = half * 8;

    v8f acc = {};
    for (int k0 = 0; k0 < K; k0 += 64) {
        __builtin_prefetch(Wp + k0 + 512, 0, 1); // global_prefetch_b8
        {
            const int ka = k0 + koff, kb = ka + 16;
            const v16bf a = load_a16(Ap, ka, kb);
            const v16bf b = load_b16(Wp, ka, kb);
            acc = __builtin_amdgcn_wmma_f32_16x16x32_bf16(
                      false, a, false, b, (short)0, acc, false, false);
        }
        {
            const int ka = k0 + 32 + koff, kb = ka + 16;
            const v16bf a = load_a16(Ap, ka, kb);
            const v16bf b = load_b16(Wp, ka, kb);
            acc = __builtin_amdgcn_wmma_f32_16x16x32_bf16(
                      false, a, false, b, (short)0, acc, false, false);
        }
    }

    // C layout: VGPR r -> lanes 0-15: (M=m0+r, N=n0+lane); lanes 16-31: (M=m0+8+r, N=n0+lane-16)
    const int cm = m0 + half * 8;
    const int cn = n0 + l;
    if (cn < N) {
        float bv = bias ? bias[cn] : 0.0f;
#pragma unroll
        for (int r = 0; r < 8; ++r) {
            const int m = cm + r;
            if (m < M) {
                float v = acc[r] + bv;
                if (relu) v = fmaxf(v, 0.0f);
                if (dec_lens) v *= (t < dec_lens[m]) ? 1.0f : 0.0f;
                C[(long long)m * ldc + cn] = v;
            }
        }
    }
}

// ---------------------------------------------------------------------------
// Setup: stable descending argsort of caption_lengths (B=64, rank-based)
// ---------------------------------------------------------------------------
__global__ void k_sort(const int* __restrict__ lens,
                       int* __restrict__ ws_order, int* __restrict__ ws_dlen,
                       int* __restrict__ out_order, int* __restrict__ out_dlen)
{
    __shared__ int ord[B_];
    const int i = threadIdx.x;
    const int li = lens[i];
    int r = 0;
    for (int j = 0; j < B_; ++j) {
        const int lj = lens[j];
        r += (lj > li) || (lj == li && j < i);
    }
    ord[r] = i;
    __syncthreads();
    const int o = ord[i];
    ws_order[i]  = o;  out_order[i] = o;
    const int dl = lens[o] - 1;
    ws_dlen[i]   = dl; out_dlen[i]  = dl;
}

__global__ void k_gather_caps(const int* __restrict__ caps, const int* __restrict__ order,
                              int* __restrict__ ws_caps, int* __restrict__ out_caps)
{
    const int b = blockIdx.x, m = threadIdx.x;  // blockDim = MAXLEN_
    const int v = caps[order[b] * MAXLEN_ + m];
    ws_caps[b * MAXLEN_ + m]  = v;
    out_caps[b * MAXLEN_ + m] = v;
}

__global__ void k_gather_enc(const float* __restrict__ enc, const int* __restrict__ order,
                             float* __restrict__ enc_s)
{
    const int b = blockIdx.x;
    const long long src = (long long)order[b] * P_ * ENC_;
    const long long dst = (long long)b * P_ * ENC_;
    for (int i = threadIdx.x * 4; i < P_ * ENC_; i += blockDim.x * 4)
        *(v4f*)(enc_s + dst + i) = *(const v4f*)(enc + src + i);
}

__global__ void k_cvt_bf16(const float* __restrict__ src, __bf16* __restrict__ dst, int n)
{
    const int i = (blockIdx.x * blockDim.x + threadIdx.x) * 8;
    if (i < n) {
        const v4f s0 = *(const v4f*)(src + i);
        const v4f s1 = *(const v4f*)(src + i + 4);
        const v8f s  = __builtin_shufflevector(s0, s1, 0, 1, 2, 3, 4, 5, 6, 7);
        *(v8bf*)(dst + i) = __builtin_convertvector(s, v8bf);
    }
}

__global__ void k_zero(float* __restrict__ p, int n)
{
    const int i = blockIdx.x * blockDim.x + threadIdx.x;
    if (i < n) p[i] = 0.0f;
}

// ---------------------------------------------------------------------------
// Per-step attention: one workgroup per (sorted) sample.
// ---------------------------------------------------------------------------
__global__ void k_attn(const float* __restrict__ enc_s, const float* __restrict__ att1,
                       const float* __restrict__ h2,
                       const float* __restrict__ dec_att_w, const float* __restrict__ dec_att_b,
                       const float* __restrict__ full_att_w, const float* __restrict__ full_att_b,
                       const float* __restrict__ gate_w1, const float* __restrict__ gate_b1,
                       const float* __restrict__ gate_w2, const float* __restrict__ gate_b2,
                       const float* __restrict__ emb, const int* __restrict__ ws_caps,
                       const int* __restrict__ ws_dlen,
                       float* __restrict__ gin, float* __restrict__ alphas_out, int t)
{
    __shared__ float att2[ATT_];
    __shared__ float ghid[ATT_];
    __shared__ float ebuf[P_];
    __shared__ float red[256];
    __shared__ float sval[4];

    const int b = blockIdx.x, tid = threadIdx.x;
    const float* h2b = h2 + b * DEC_;
    const float maskf = (t < ws_dlen[b]) ? 1.0f : 0.0f;

    {
        const int d = tid & (ATT_ - 1);
        const float* w = (tid < ATT_) ? (dec_att_w + d * DEC_) : (gate_w1 + d * DEC_);
        float s = 0.0f;
        for (int k = 0; k < DEC_; k += 4) {
            const v4f hv = *(const v4f*)(h2b + k);
            const v4f wv = *(const v4f*)(w + k);
            s += hv.x * wv.x + hv.y * wv.y + hv.z * wv.z + hv.w * wv.w;
        }
        if (tid < ATT_) att2[d] = s + dec_att_b[d];
        else            ghid[d] = fmaxf(s + gate_b1[d], 0.0f);
    }
    __syncthreads();

    float ev = -1e30f;
    if (tid < P_) {
        const float* a1 = att1 + ((long long)b * P_ + tid) * ATT_;
        float s = full_att_b[0];
        for (int a = 0; a < ATT_; ++a)
            s += fmaxf(a1[a] + att2[a], 0.0f) * full_att_w[a];
        ebuf[tid] = s;
        ev = s;
    }
    __syncthreads();

    // softmax max
    red[tid] = ev;
    __syncthreads();
    for (int off = 128; off > 0; off >>= 1) {
        if (tid < off) red[tid] = fmaxf(red[tid], red[tid + off]);
        __syncthreads();
    }
    if (tid == 0) sval[0] = red[0];
    __syncthreads();
    // softmax sum
    const float ex = (tid < P_) ? expf(ev - sval[0]) : 0.0f;
    red[tid] = ex;
    __syncthreads();
    for (int off = 128; off > 0; off >>= 1) {
        if (tid < off) red[tid] += red[tid + off];
        __syncthreads();
    }
    if (tid == 0) sval[1] = red[0];
    __syncthreads();
    if (tid < P_) {
        const float a = ex / sval[1];
        ebuf[tid] = a;
        alphas_out[((long long)b * T_ + t) * P_ + tid] = a * maskf;
    }
    __syncthreads();

    // gate scalar
    red[tid] = (tid < ATT_) ? ghid[tid] * gate_w2[tid] : 0.0f;
    __syncthreads();
    for (int off = 128; off > 0; off >>= 1) {
        if (tid < off) red[tid] += red[tid + off];
        __syncthreads();
    }
    if (tid == 0) sval[2] = 1.0f / (1.0f + expf(-(red[0] + gate_b2[0])));
    __syncthreads();
    const float gate = sval[2];

    // gin = [emb(token), gate * sum_p alpha_p * enc[b,p,:]]
    const int tok = ws_caps[b * MAXLEN_ + t];
    if (tid < EMB_)
        gin[b * (EMB_ + ENC_) + tid] = emb[(long long)tok * EMB_ + tid];
    const float* eb = enc_s + (long long)b * P_ * ENC_;
    for (int d = tid; d < ENC_; d += 256) {
        float s = 0.0f;
        for (int p = 0; p < P_; ++p) s += ebuf[p] * eb[(long long)p * ENC_ + d];
        gin[b * (EMB_ + ENC_) + EMB_ + d] = s * gate;
    }
}

// ---------------------------------------------------------------------------
// GRU gate nonlinearity + masked state commit (hcur untouched when mask==0)
// ---------------------------------------------------------------------------
__global__ void k_gru(const float* __restrict__ gi, const float* __restrict__ gh,
                      float* __restrict__ hcur, float* __restrict__ hnew,
                      const int* __restrict__ ws_dlen, int t)
{
    const int idx = blockIdx.x * blockDim.x + threadIdx.x; // over B_*DEC_
    const int b = idx >> 9, d = idx & (DEC_ - 1);
    const int gbase = b * 3 * DEC_ + d;
    const float ir = gi[gbase], iz = gi[gbase + DEC_], ig = gi[gbase + 2 * DEC_];
    const float hr = gh[gbase], hz = gh[gbase + DEC_], hg = gh[gbase + 2 * DEC_];
    const float r = 1.0f / (1.0f + expf(-(ir + hr)));
    const float z = 1.0f / (1.0f + expf(-(iz + hz)));
    const float n = tanhf(ig + r * hg);
    const float hprev = hcur[idx];
    const float h = (1.0f - z) * n + z * hprev;
    hnew[idx] = h;
    hcur[idx] = (t < ws_dlen[b]) ? h : hprev;
}

// out = LN(x1 + x2) * g + beta ; one block (256 threads) per row of 512
__global__ void k_add_ln(const float* __restrict__ x1, const float* __restrict__ x2,
                         const float* __restrict__ g, const float* __restrict__ beta,
                         float* __restrict__ out)
{
    __shared__ float s1[256], s2[256];
    const int b = blockIdx.x, tid = threadIdx.x;
    const float a0 = x1[b * DEC_ + tid]       + x2[b * DEC_ + tid];
    const float a1 = x1[b * DEC_ + 256 + tid] + x2[b * DEC_ + 256 + tid];
    s1[tid] = a0 + a1;
    s2[tid] = a0 * a0 + a1 * a1;
    __syncthreads();
    for (int off = 128; off > 0; off >>= 1) {
        if (tid < off) { s1[tid] += s1[tid + off]; s2[tid] += s2[tid + off]; }
        __syncthreads();
    }
    const float m   = s1[0] * (1.0f / DEC_);
    const float var = s2[0] * (1.0f / DEC_) - m * m;
    const float inv = rsqrtf(var + EPS_);
    out[b * DEC_ + tid]       = (a0 - m) * inv * g[tid]       + beta[tid];
    out[b * DEC_ + 256 + tid] = (a1 - m) * inv * g[256 + tid] + beta[256 + tid];
}

// ---------------------------------------------------------------------------
extern "C" void kernel_launch(void* const* d_in, const int* in_sizes, int n_in,
                              void* d_out, int out_size, void* d_ws, size_t ws_size,
                              hipStream_t stream)
{
    const float* encoder_out = (const float*)d_in[0];
    const int*   caps        = (const int*)  d_in[1];
    const int*   lens        = (const int*)  d_in[2];
    const float* enc_att_w   = (const float*)d_in[3];
    const float* enc_att_b   = (const float*)d_in[4];
    const float* dec_att_w   = (const float*)d_in[5];
    const float* dec_att_b   = (const float*)d_in[6];
    const float* full_att_w  = (const float*)d_in[7];
    const float* full_att_b  = (const float*)d_in[8];
    const float* gate_w1     = (const float*)d_in[9];
    const float* gate_b1     = (const float*)d_in[10];
    const float* gate_w2     = (const float*)d_in[11];
    const float* gate_b2     = (const float*)d_in[12];
    const float* emb         = (const float*)d_in[13];
    const float* gru1_wih    = (const float*)d_in[14];
    const float* gru1_whh    = (const float*)d_in[15];
    const float* gru1_bih    = (const float*)d_in[16];
    const float* gru1_bhh    = (const float*)d_in[17];
    const float* gru2_wih    = (const float*)d_in[18];
    const float* gru2_whh    = (const float*)d_in[19];
    const float* gru2_bih    = (const float*)d_in[20];
    const float* gru2_bhh    = (const float*)d_in[21];
    const float* proj_w      = (const float*)d_in[22];
    const float* proj_b      = (const float*)d_in[23];
    const float* ln1_g       = (const float*)d_in[24];
    const float* ln1_b       = (const float*)d_in[25];
    const float* ln2_g       = (const float*)d_in[26];
    const float* ln2_b       = (const float*)d_in[27];
    const float* ffn_w1      = (const float*)d_in[28];
    const float* ffn_b1      = (const float*)d_in[29];
    const float* ffn_w2      = (const float*)d_in[30];
    const float* ffn_b2      = (const float*)d_in[31];

    // ---- d_out layout (flat, return order) ----
    float* out_pred  = (float*)d_out;                                  // B*T*V
    int*   out_caps  = (int*)(out_pred + (size_t)B_ * T_ * V_);        // B*MAXLEN
    int*   out_dlen  = out_caps + (size_t)B_ * MAXLEN_;                // B
    float* out_alpha = (float*)(out_dlen + B_);                        // B*T*P
    int*   out_order = (int*)(out_alpha + (size_t)B_ * T_ * P_);       // B

    // ---- workspace layout ----
    char* wsp = (char*)d_ws;
    auto alloc = [&](size_t bytes) {
        void* p = wsp;
        wsp += (bytes + 255) & ~(size_t)255;
        return p;
    };
    int*    ws_order = (int*)   alloc(B_ * sizeof(int));
    int*    ws_dlen  = (int*)   alloc(B_ * sizeof(int));
    int*    ws_caps  = (int*)   alloc((size_t)B_ * MAXLEN_ * sizeof(int));
    float*  enc_s    = (float*) alloc((size_t)B_ * P_ * ENC_ * sizeof(float));
    float*  att1     = (float*) alloc((size_t)B_ * P_ * ATT_ * sizeof(float));
    float*  h1cur    = (float*) alloc((size_t)B_ * DEC_ * sizeof(float));
    float*  h2cur    = (float*) alloc((size_t)B_ * DEC_ * sizeof(float));
    float*  h1new    = (float*) alloc((size_t)B_ * DEC_ * sizeof(float));
    float*  h2new    = (float*) alloc((size_t)B_ * DEC_ * sizeof(float));
    float*  h1res    = (float*) alloc((size_t)B_ * DEC_ * sizeof(float));
    float*  h2res    = (float*) alloc((size_t)B_ * DEC_ * sizeof(float));
    float*  gin      = (float*) alloc((size_t)B_ * (EMB_ + ENC_) * sizeof(float));
    float*  gbuf_i   = (float*) alloc((size_t)B_ * 3 * DEC_ * sizeof(float));
    float*  gbuf_h   = (float*) alloc((size_t)B_ * 3 * DEC_ * sizeof(float));
    float*  projbuf  = (float*) alloc((size_t)B_ * DEC_ * sizeof(float));
    float*  fbuf     = (float*) alloc((size_t)B_ * FFN_ * sizeof(float));
    __bf16* bw_enc   = (__bf16*)alloc((size_t)ATT_ * ENC_ * sizeof(__bf16));
    __bf16* bw_g1i   = (__bf16*)alloc((size_t)3 * DEC_ * (EMB_ + ENC_) * sizeof(__bf16));
    __bf16* bw_g1h   = (__bf16*)alloc((size_t)3 * DEC_ * DEC_ * sizeof(__bf16));
    __bf16* bw_g2i   = (__bf16*)alloc((size_t)3 * DEC_ * DEC_ * sizeof(__bf16));
    __bf16* bw_g2h   = (__bf16*)alloc((size_t)3 * DEC_ * DEC_ * sizeof(__bf16));
    __bf16* bw_proj  = (__bf16*)alloc((size_t)DEC_ * (EMB_ + ENC_) * sizeof(__bf16));
    __bf16* bw_f1    = (__bf16*)alloc((size_t)FFN_ * DEC_ * sizeof(__bf16));
    __bf16* bw_f2    = (__bf16*)alloc((size_t)V_ * FFN_ * sizeof(__bf16));

    // ---- setup ----
    k_sort<<<1, B_, 0, stream>>>(lens, ws_order, ws_dlen, out_order, out_dlen);
    k_gather_caps<<<B_, MAXLEN_, 0, stream>>>(caps, ws_order, ws_caps, out_caps);
    k_gather_enc<<<B_, 256, 0, stream>>>(encoder_out, ws_order, enc_s);

    auto cvt = [&](const float* s, __bf16* d, int n) {  // n % 8 == 0 at all call sites
        k_cvt_bf16<<<(n / 8 + 255) / 256, 256, 0, stream>>>(s, d, n);
    };
    cvt(enc_att_w, bw_enc,  ATT_ * ENC_);
    cvt(gru1_wih,  bw_g1i,  3 * DEC_ * (EMB_ + ENC_));
    cvt(gru1_whh,  bw_g1h,  3 * DEC_ * DEC_);
    cvt(gru2_wih,  bw_g2i,  3 * DEC_ * DEC_);
    cvt(gru2_whh,  bw_g2h,  3 * DEC_ * DEC_);
    cvt(proj_w,    bw_proj, DEC_ * (EMB_ + ENC_));
    cvt(ffn_w1,    bw_f1,   FFN_ * DEC_);
    cvt(ffn_w2,    bw_f2,   V_ * FFN_);

    k_zero<<<(B_ * DEC_ + 255) / 256, 256, 0, stream>>>(h1cur, B_ * DEC_);
    k_zero<<<(B_ * DEC_ + 255) / 256, 256, 0, stream>>>(h2cur, B_ * DEC_);

    auto gemm = [&](const float* A, int M, int lda, const __bf16* W, int N, int K,
                    float* C, long long ldc, const float* bias, int relu,
                    const int* mask, int t) {
        dim3 grid((N + 63) / 64, (M + 15) / 16);
        gemm_bf16_wmma<<<grid, 128, 0, stream>>>(A, M, lda, W, N, K, C, ldc,
                                                 bias, relu, mask, t);
    };

    // att1 = enc_s @ enc_att_w^T + b  (M = B*P = 12544, N = 128, K = 512)
    gemm(enc_s, B_ * P_, ENC_, bw_enc, ATT_, ENC_, att1, ATT_, enc_att_b, 0, nullptr, 0);

    // ---- sequential decode ----
    for (int t = 0; t < T_; ++t) {
        k_attn<<<B_, 256, 0, stream>>>(enc_s, att1, h2cur,
                                       dec_att_w, dec_att_b, full_att_w, full_att_b,
                                       gate_w1, gate_b1, gate_w2, gate_b2,
                                       emb, ws_caps, ws_dlen, gin, out_alpha, t);

        // GRU1
        gemm(gin,   B_, EMB_ + ENC_, bw_g1i, 3 * DEC_, EMB_ + ENC_, gbuf_i, 3 * DEC_, gru1_bih, 0, nullptr, 0);
        gemm(h1cur, B_, DEC_,        bw_g1h, 3 * DEC_, DEC_,        gbuf_h, 3 * DEC_, gru1_bhh, 0, nullptr, 0);
        k_gru<<<(B_ * DEC_) / 256, 256, 0, stream>>>(gbuf_i, gbuf_h, h1cur, h1new, ws_dlen, t);

        // h1_res = LN(h1_new + gin @ proj_w^T + proj_b)
        gemm(gin, B_, EMB_ + ENC_, bw_proj, DEC_, EMB_ + ENC_, projbuf, DEC_, proj_b, 0, nullptr, 0);
        k_add_ln<<<B_, 256, 0, stream>>>(h1new, projbuf, ln1_g, ln1_b, h1res);

        // GRU2
        gemm(h1res, B_, DEC_, bw_g2i, 3 * DEC_, DEC_, gbuf_i, 3 * DEC_, gru2_bih, 0, nullptr, 0);
        gemm(h2cur, B_, DEC_, bw_g2h, 3 * DEC_, DEC_, gbuf_h, 3 * DEC_, gru2_bhh, 0, nullptr, 0);
        k_gru<<<(B_ * DEC_) / 256, 256, 0, stream>>>(gbuf_i, gbuf_h, h2cur, h2new, ws_dlen, t);

        // h2_res = LN(h2_new + h1_res)
        k_add_ln<<<B_, 256, 0, stream>>>(h2new, h1res, ln2_g, ln2_b, h2res);

        // FFN: preds[b,t,:] = relu(h2_res@W1^T+b1)@W2^T + b2, masked
        gemm(h2res, B_, DEC_, bw_f1, FFN_, DEC_, fbuf, FFN_, ffn_b1, 1, nullptr, 0);
        gemm(fbuf,  B_, FFN_, bw_f2, V_,   FFN_, out_pred + (long long)t * V_,
             (long long)T_ * V_, ffn_b2, 0, ws_dlen, t);
    }
}
